// TestAttentionStaticQuantPatternModel_80410377715848
// MI455X (gfx1250) — 8-bit matrix operand order repaired and checked by reading, compiled, not run on hardware
//
#include <hip/hip_runtime.h>
#include <hip/hip_bf16.h>

typedef int   v16i __attribute__((ext_vector_type(16)));
typedef float v8f  __attribute__((ext_vector_type(8)));
typedef float v4f  __attribute__((ext_vector_type(4)));
typedef int   v4i  __attribute__((ext_vector_type(4)));

#define HKV 8
#define GQA 4
#define HD  128
#define BSZ 16
#define MAXB 64
#define CH  64
#define HID 4096

#if defined(__has_builtin)
#if __has_builtin(__builtin_amdgcn_cvt_pk_fp8_f32)
#define HAVE_HW_FP8 1
#endif
#if __has_builtin(__builtin_amdgcn_global_load_async_to_lds_b128)
#define HAVE_ASYNC_LDS 1
#endif
#endif

#ifdef HAVE_ASYNC_LDS
typedef __attribute__((address_space(1))) v4i gv4i;
typedef __attribute__((address_space(3))) v4i lv4i;
#endif

#ifdef HAVE_HW_FP8
template <bool HI>
static __device__ __forceinline__ int pk_fp8(float a, float b, int old) {
  return __builtin_amdgcn_cvt_pk_fp8_f32(a, b, old, HI);
}
#else
static __device__ __forceinline__ unsigned sw_fp8(float x) {
  union { float f; unsigned u; } cv; cv.f = x;
  unsigned s = (cv.u >> 24) & 0x80u;
  float ax = fabsf(x);
  if (!(ax >= 1.953125e-3f)) {
    int q = (int)(ax * 512.0f + 0.5f);
    return s | (unsigned)q;
  }
  cv.f = ax;
  int e = (int)((cv.u >> 23) & 0xff) - 127;
  unsigned m3 = ((cv.u & 0x7fffffu) + 0x80000u) >> 20;
  if (m3 == 8u) { m3 = 0u; e += 1; }
  if (e > 8) { e = 8; m3 = 6u; }
  return s | ((unsigned)(e + 7) << 3) | m3;
}
template <bool HI>
static __device__ __forceinline__ int pk_fp8(float a, float b, int old) {
  unsigned v2 = sw_fp8(a) | (sw_fp8(b) << 8);
  return HI ? (int)(((unsigned)old & 0xffffu) | (v2 << 16))
            : (int)(((unsigned)old & 0xffff0000u) | v2);
}
#endif

static __device__ __forceinline__ float clamp8(float x) {
  return fminf(fmaxf(x, -448.0f), 448.0f);
}

static __device__ __forceinline__ void wait_async0() {
#ifdef HAVE_ASYNC_LDS
#if __has_builtin(__builtin_amdgcn_s_wait_asynccnt)
  __builtin_amdgcn_s_wait_asynccnt(0);
#else
  asm volatile("s_wait_asynccnt 0" ::: "memory");
#endif
#endif
}

__global__ __launch_bounds__(256) void attn_kernel(
    const float* __restrict__ q, const float* __restrict__ knew,
    const float* __restrict__ vnew, const float* __restrict__ cache,
    const int* __restrict__ btab, const int* __restrict__ slen,
    const float* __restrict__ scale, unsigned char* __restrict__ xq)
{
  const int h = blockIdx.x;
  const int b = blockIdx.y;
  const int t = threadIdx.x;

  __shared__ float qs[GQA * HD];
  __shared__ float vs[CH * HD];
  __shared__ float partl[4 * GQA * CH];
  __shared__ float plds[GQA * CH];
  __shared__ float mrun[GQA], lrun[GQA], alph[GQA];
  __shared__ float wred[8];

  const int len = slen[b];

  {
    const float* qp = q + (size_t)b * HID + (size_t)h * (GQA * HD);
    qs[t] = qp[t];
    qs[t + 256] = qp[t + 256];
  }
  if (t < GQA) { mrun[t] = -3.0e38f; lrun[t] = 0.0f; }
  __syncthreads();

  const int g_o  = t >> 6;
  const int sl   = t & 63;
  const int part = t >> 6;
  const int wv   = t >> 5;
  const int lane = t & 31;
  const int d2   = (t & 63) * 2;

  float o0 = 0.0f, o1 = 0.0f;

  const int nch = (len + CH - 1) / CH;
  for (int c = 0; c < nch; ++c) {
    __syncthreads();

    #pragma unroll
    for (int r = 0; r < 8; ++r) {
      const int j  = r * 1024 + t * 4;
      const int vr = j >> 7;
      const int vd = j & 127;
      const int sg = c * CH + vr;
      if (sg < len) {
        const float* src;
        if (sg == len - 1) {
          src = vnew + (size_t)b * (HKV * HD) + h * HD + vd;
        } else {
          const int blk = btab[b * MAXB + (sg >> 4)];
          src = cache + ((size_t)(blk * 2 + 1) * BSZ + (sg & 15)) * (HKV * HD) +
                (size_t)h * HD + vd;
        }
#ifdef HAVE_ASYNC_LDS
        __builtin_amdgcn_global_load_async_to_lds_b128(
            (gv4i*)src, (lv4i*)(vs + j), 0, 0);
#else
        *(v4f*)(vs + j) = __builtin_nontemporal_load((const v4f*)src);
#endif
      } else {
        const v4f z = {0.f, 0.f, 0.f, 0.f};
        *(v4f*)(vs + j) = z;
      }
    }

    const int s = c * CH + sl;
    const bool valid = (s < len);
    float a0 = 0.f, a1 = 0.f, a2 = 0.f, a3 = 0.f;
    if (valid) {
      const float* kp;
      if (s == len - 1) {
        kp = knew + (size_t)b * (HKV * HD) + h * HD;
      } else {
        const int blk = btab[b * MAXB + (s >> 4)];
        kp = cache + ((size_t)(blk * 2) * BSZ + (s & 15)) * (HKV * HD) + (size_t)h * HD;
      }
      #pragma unroll
      for (int i = 0; i < 8; ++i) {
        const int dd = part * 32 + i * 4;
        const v4f kv4 = __builtin_nontemporal_load((const v4f*)(kp + dd));
        const float* q0 = qs + dd;
        a0 += q0[0] * kv4.x + q0[1] * kv4.y + q0[2] * kv4.z + q0[3] * kv4.w;
        const float* q1 = q0 + HD;
        a1 += q1[0] * kv4.x + q1[1] * kv4.y + q1[2] * kv4.z + q1[3] * kv4.w;
        const float* q2 = q1 + HD;
        a2 += q2[0] * kv4.x + q2[1] * kv4.y + q2[2] * kv4.z + q2[3] * kv4.w;
        const float* q3 = q2 + HD;
        a3 += q3[0] * kv4.x + q3[1] * kv4.y + q3[2] * kv4.z + q3[3] * kv4.w;
      }
    }
    partl[(part * GQA + 0) * CH + sl] = a0;
    partl[(part * GQA + 1) * CH + sl] = a1;
    partl[(part * GQA + 2) * CH + sl] = a2;
    partl[(part * GQA + 3) * CH + sl] = a3;
    wait_async0();
    __syncthreads();

    float sc = partl[(0 * GQA + g_o) * CH + sl] + partl[(1 * GQA + g_o) * CH + sl] +
               partl[(2 * GQA + g_o) * CH + sl] + partl[(3 * GQA + g_o) * CH + sl];
    sc *= 0.08838834764831845f;
    if (!valid) sc = -3.0e38f;

    float wm = sc;
    #pragma unroll
    for (int m = 16; m >= 1; m >>= 1) wm = fmaxf(wm, __shfl_xor(wm, m, 32));
    if (lane == 0) wred[wv] = wm;
    __syncthreads();
    const float cmax  = fmaxf(wred[wv], wred[wv ^ 1]);
    const float m_old = mrun[g_o];
    const float m_new = fmaxf(m_old, cmax);
    const float p = valid ? __expf(sc - m_new) : 0.0f;
    plds[g_o * CH + sl] = p;
    __syncthreads();

    float wsum = p;
    #pragma unroll
    for (int m = 16; m >= 1; m >>= 1) wsum += __shfl_xor(wsum, m, 32);
    if (lane == 0) wred[wv] = wsum;
    __syncthreads();
    const float psum = wred[wv] + wred[wv ^ 1];
    if ((t & 63) == 0) {
      const float alpha = __expf(m_old - m_new);
      lrun[g_o] = lrun[g_o] * alpha + psum;
      mrun[g_o] = m_new;
      alph[g_o] = alpha;
    }
    __syncthreads();

    const float alpha = alph[g_o];
    o0 *= alpha; o1 *= alpha;
    const float* prow = plds + g_o * CH;
    const float* vcol = vs + d2;
    #pragma unroll 4
    for (int si = 0; si < CH; ++si) {
      const float pv = prow[si];
      const float2 vv = *(const float2*)(vcol + si * HD);
      o0 += pv * vv.x;
      o1 += pv * vv.y;
    }
  }

  __syncthreads();
  const float linv  = 1.0f / lrun[g_o];
  const float inv_s = 1.0f / scale[0];
  const float x0 = clamp8(o0 * linv * inv_s);
  const float x1 = clamp8(o1 * linv * inv_s);
  const int   pk = pk_fp8<false>(x0, x1, 0);
  *(unsigned short*)(xq + (size_t)b * HID + h * (GQA * HD) + g_o * HD + d2) =
      (unsigned short)(pk & 0xffff);
}

__global__ __launch_bounds__(128) void gemm_kernel(
    const unsigned char* __restrict__ xq, const float* __restrict__ w,
    const float* __restrict__ wscale_p, const float* __restrict__ scale_p,
    float* __restrict__ out)
{
  const int lane  = threadIdx.x & 31;
  const int nTile = blockIdx.x * 4 + (threadIdx.x >> 5);
  const bool loh  = (lane < 16);
  const int  n    = nTile * 16 + (lane & 15);

  const float* wrow = w + (size_t)n * HID;
  const unsigned int* xr0 = (const unsigned int*)(xq + (size_t)(0 * 16 + (lane & 15)) * HID);
  const unsigned int* xr1 = (const unsigned int*)(xq + (size_t)(1 * 16 + (lane & 15)) * HID);
  const unsigned int* xr2 = (const unsigned int*)(xq + (size_t)(2 * 16 + (lane & 15)) * HID);
  const unsigned int* xr3 = (const unsigned int*)(xq + (size_t)(3 * 16 + (lane & 15)) * HID);

  const float inv_ws = 1.0f / wscale_p[0];
  v8f acc0 = {0,0,0,0,0,0,0,0};
  v8f acc1 = {0,0,0,0,0,0,0,0};
  v8f acc2 = {0,0,0,0,0,0,0,0};
  v8f acc3 = {0,0,0,0,0,0,0,0};

  const int aoff = loh ? 0 : 4;
  const int boff = loh ? 0 : 16;

  for (int ks = 0; ks < 32; ++ks) {
    const float* wk = wrow + ks * 128 + boff;
    __builtin_prefetch(wrow + (ks + 1) * 128, 0, 1);

    v16i bf;
    #pragma unroll
    for (int c = 0; c < 4; ++c) {
      #pragma unroll
      for (int r = 0; r < 4; ++r) {
        const v4f f = __builtin_nontemporal_load((const v4f*)(wk + c * 32 + r * 4));
        const float b0 = clamp8(f.x * inv_ws);
        const float b1 = clamp8(f.y * inv_ws);
        const float b2 = clamp8(f.z * inv_ws);
        const float b3 = clamp8(f.w * inv_ws);
        int d = pk_fp8<false>(b0, b1, 0);
        d = pk_fp8<true>(b2, b3, d);
        bf[c * 4 + r] = d;
      }
    }

    const int ab = ks * 32 + aoff;
    v16i af0, af1, af2, af3;
    #pragma unroll
    for (int j = 0; j < 8; ++j) {
      uint2 p;
      p = *(const uint2*)(xr0 + ab + (j >> 1) * 8 + (j & 1) * 2); af0[2 * j] = (int)p.x; af0[2 * j + 1] = (int)p.y;
      p = *(const uint2*)(xr1 + ab + (j >> 1) * 8 + (j & 1) * 2); af1[2 * j] = (int)p.x; af1[2 * j + 1] = (int)p.y;
      p = *(const uint2*)(xr2 + ab + (j >> 1) * 8 + (j & 1) * 2); af2[2 * j] = (int)p.x; af2[2 * j + 1] = (int)p.y;
      p = *(const uint2*)(xr3 + ab + (j >> 1) * 8 + (j & 1) * 2); af3[2 * j] = (int)p.x; af3[2 * j + 1] = (int)p.y;
    }

    acc0 = __builtin_amdgcn_wmma_f32_16x16x128_fp8_fp8(af0, bf, (short)0, acc0, false, false);
    acc1 = __builtin_amdgcn_wmma_f32_16x16x128_fp8_fp8(af1, bf, (short)0, acc1, false, false);
    acc2 = __builtin_amdgcn_wmma_f32_16x16x128_fp8_fp8(af2, bf, (short)0, acc2, false, false);
    acc3 = __builtin_amdgcn_wmma_f32_16x16x128_fp8_fp8(af3, bf, (short)0, acc3, false, false);
  }

  const float sc = scale_p[0] * wscale_p[0];
  const int mofs = loh ? 0 : 8;
  #pragma unroll
  for (int r = 0; r < 8; ++r) {
    out[(size_t)(0 * 16 + r + mofs) * HID + n] = acc0[r] * sc;
    out[(size_t)(1 * 16 + r + mofs) * HID + n] = acc1[r] * sc;
    out[(size_t)(2 * 16 + r + mofs) * HID + n] = acc2[r] * sc;
    out[(size_t)(3 * 16 + r + mofs) * HID + n] = acc3[r] * sc;
  }
}

extern "C" void kernel_launch(void* const* d_in, const int* in_sizes, int n_in,
                              void* d_out, int out_size, void* d_ws, size_t ws_size,
                              hipStream_t stream) {
  (void)in_sizes; (void)n_in; (void)out_size; (void)ws_size;
  const float* q      = (const float*)d_in[0];
  const float* k      = (const float*)d_in[1];
  const float* v      = (const float*)d_in[2];
  const float* cache  = (const float*)d_in[3];
  const float* w      = (const float*)d_in[4];
  const float* wscale = (const float*)d_in[5];
  const float* scale  = (const float*)d_in[6];
  const int*   btab   = (const int*)d_in[7];
  const int*   slen   = (const int*)d_in[8];
  float* out          = (float*)d_out;
  unsigned char* xq   = (unsigned char*)d_ws;

  attn_kernel<<<dim3(HKV, 64), 256, 0, stream>>>(q, k, v, cache, btab, slen, scale, xq);
  gemm_kernel<<<64, 128, 0, stream>>>(xq, w, wscale, scale, out);
}
